// DeformConv1d_90984587198677
// MI455X (gfx1250) — matrix operand order repaired and checked by reading, not compiled, not run on hardware
//
#include <hip/hip_runtime.h>

typedef __attribute__((ext_vector_type(16))) __bf16 v16bf;
typedef __attribute__((ext_vector_type(8)))  float  v8f;

#define WIN    4096
#define CCH    256
#define OCH    256
#define KTAP   3
#define CK     (CCH * KTAP)   // 768
#define NTILE  32
#define CHUNK  32

__device__ __forceinline__ unsigned pack2bf(float lo, float hi) {
  unsigned a = __float_as_uint(lo);
  unsigned b = __float_as_uint(hi);
  a = (a + 0x7FFFu + ((a >> 16) & 1u)) >> 16;   // round-to-nearest-even
  b = (b + 0x7FFFu + ((b >> 16) & 1u)) >> 16;
  return a | (b << 16);
}

struct Coef { int i0; int i1; float w0; float w1; };

union FragBF  { unsigned u[8]; v16bf v; };
union FragAcc { v8f v; float f[8]; };

// ---- prologue: weight f32 -> bf16 (same [oc][768] element order) ----
__global__ __launch_bounds__(256)
void convert_weight_bf16(const float* __restrict__ w, unsigned* __restrict__ o) {
  const int i = blockIdx.x * 256 + threadIdx.x;     // one packed dword per thread
  const float2 v = ((const float2*)w)[i];
  o[i] = pack2bf(v.x, v.y);
}

template <bool PRECONV>
__global__ __launch_bounds__(256)
void deform_conv1d_wmma(const float* __restrict__ x,
                        const float* __restrict__ weight,
                        const unsigned short* __restrict__ wbf,
                        const float* __restrict__ offset,
                        const float* __restrict__ mask,
                        const float* __restrict__ bias,
                        float* __restrict__ out) {
  __shared__ Coef coef[KTAP * NTILE];                 // 96 * 16B
  __shared__ __align__(16) unsigned ldsB[2][512];     // 2 bufs x (2 halves x 32 lanes x 8 dw)

  const int tid  = threadIdx.x;
  const int lane = tid & 31;
  const int wave = tid >> 5;

  const int tilesPerB = WIN / NTILE;                  // 128
  const int bIdx = blockIdx.x / tilesPerB;
  const int w0   = (blockIdx.x % tilesPerB) * NTILE;

  // ---- interpolation coefficient table (OOB folded into zero weights) ----
  if (tid < KTAP * NTILE) {
    const int k = tid / NTILE;
    const int n = tid % NTILE;
    const int w = w0 + n;
    const int om = (bIdx * KTAP + k) * WIN + w;
    const float p  = (float)(w - 1 + k) + offset[om]; // stride=1, pad=1, dil=1
    const float pf = floorf(p);
    const int   i0 = (int)pf;
    const int   i1 = i0 + 1;
    const float t  = p - pf;
    const float mm = mask[om];
    const bool v0 = (i0 >= 0) && (i0 < WIN);
    const bool v1 = (i1 >= 0) && (i1 < WIN);
    Coef c;
    c.i0 = v0 ? i0 : 0;
    c.i1 = v1 ? i1 : 0;
    c.w0 = v0 ? (1.0f - t) * mm : 0.0f;
    c.w1 = v1 ? t * mm : 0.0f;
    coef[tid] = c;
  }
  __syncthreads();

  // B-producer coordinates: thread (wave=dword slot, lane) fills one packed dword per half
  const int vB    = wave;                 // packed-dword (VGPR) slot 0..7
  const int nIn16 = lane & 15;            // N within 16-wide half
  const int kb    = (lane >> 4) << 3;     // lanes 0-15: K 0-7/16-23, lanes 16-31: K 8-15/24-31
  const int rl0   = kb + 2 * vB + ((vB >> 2) << 3);  // K pair base within chunk

  const float* xb = x + (size_t)bIdx * CCH * WIN;

  FragAcc acc[2][2];
  #pragma unroll
  for (int i = 0; i < 2; ++i)
    #pragma unroll
    for (int j = 0; j < 2; ++j)
      #pragma unroll
      for (int q = 0; q < 8; ++q) acc[i][j].f[q] = 0.0f;

  const int m0       = wave * 32;         // this wave's 32 OC rows
  const int ocInTile = lane & 15;
  const int rABase   = (lane < 16) ? 0 : 8;  // A layout: lanes 0-15 K0-7/16-23, lanes 16-31 K8-15/24-31

  int buf = 0;
  for (int r0 = 0; r0 < CK; r0 += CHUNK) {
    // ---- produce sampled B chunk (bf16, fragment layout) into LDS ----
    #pragma unroll
    for (int half = 0; half < 2; ++half) {
      const int n = half * 16 + nIn16;
      float vv[2];
      #pragma unroll
      for (int e = 0; e < 2; ++e) {
        const int r = r0 + rl0 + e;       // reduction index = c*3 + k
        const int c = r / 3;
        const int k = r - 3 * c;
        const Coef cf = coef[k * NTILE + n];
        const float* xc = xb + (size_t)c * WIN;
        vv[e] = xc[cf.i0] * cf.w0 + xc[cf.i1] * cf.w1;
      }
      ldsB[buf][half * 256 + lane * 8 + vB] = pack2bf(vv[0], vv[1]);
    }
    __syncthreads();

    // ---- load B fragments (2x ds_load_b128 per half) ----
    FragBF bfr[2];
    #pragma unroll
    for (int half = 0; half < 2; ++half) {
      const uint4* p = (const uint4*)&ldsB[buf][half * 256 + lane * 8];
      uint4 q0 = p[0];
      uint4 q1 = p[1];
      bfr[half].u[0] = q0.x; bfr[half].u[1] = q0.y; bfr[half].u[2] = q0.z; bfr[half].u[3] = q0.w;
      bfr[half].u[4] = q1.x; bfr[half].u[5] = q1.y; bfr[half].u[6] = q1.z; bfr[half].u[7] = q1.w;
    }

    // ---- load A fragments ----
    FragBF af[2];
    if (PRECONV) {
      // bf16 weights already in fragment element order: 2x global_load_b128, zero VALU
      #pragma unroll
      for (int i = 0; i < 2; ++i) {
        const int oc = m0 + i * 16 + ocInTile;
        const uint4* wp = (const uint4*)(wbf + (size_t)oc * CK + r0 + rABase);
        uint4 g0 = wp[0];                  // K rA+0..7
        uint4 g1 = wp[2];                  // K rA+16..23 (+32 bytes)
        af[i].u[0] = g0.x; af[i].u[1] = g0.y; af[i].u[2] = g0.z; af[i].u[3] = g0.w;
        af[i].u[4] = g1.x; af[i].u[5] = g1.y; af[i].u[6] = g1.z; af[i].u[7] = g1.w;
      }
    } else {
      // fallback: load f32 weights and convert in-loop
      #pragma unroll
      for (int i = 0; i < 2; ++i) {
        const int oc = m0 + i * 16 + ocInTile;
        const float4* wp = (const float4*)(weight + (size_t)oc * CK + r0 + rABase);
        float4 a0 = wp[0];
        float4 a1 = wp[1];
        float4 a2 = wp[4];
        float4 a3 = wp[5];
        af[i].u[0] = pack2bf(a0.x, a0.y);
        af[i].u[1] = pack2bf(a0.z, a0.w);
        af[i].u[2] = pack2bf(a1.x, a1.y);
        af[i].u[3] = pack2bf(a1.z, a1.w);
        af[i].u[4] = pack2bf(a2.x, a2.y);
        af[i].u[5] = pack2bf(a2.z, a2.w);
        af[i].u[6] = pack2bf(a3.x, a3.y);
        af[i].u[7] = pack2bf(a3.z, a3.w);
      }
    }

    // ---- matrix core: 4 x v_wmma_f32_16x16x32_bf16 ----
    #pragma unroll
    for (int i = 0; i < 2; ++i)
      #pragma unroll
      for (int j = 0; j < 2; ++j)
        acc[i][j].v = __builtin_amdgcn_wmma_f32_16x16x32_bf16(
            false, af[i].v, false, bfr[j].v, (short)0, acc[i][j].v, false, false);

    buf ^= 1;   // double buffer: one barrier per chunk is sufficient
  }

  // ---- epilogue: bias add + coalesced stores (C/D layout: M=j+8*(lane/16), N=lane%16) ----
  const int mrow = (lane >> 4) << 3;
  const int ncol = lane & 15;
  #pragma unroll
  for (int i = 0; i < 2; ++i) {
    const int ocBase = m0 + i * 16 + mrow;
    const float4* bp = (const float4*)(bias + ocBase);
    float4 b0 = bp[0];
    float4 b1 = bp[1];
    float bv[8] = { b0.x, b0.y, b0.z, b0.w, b1.x, b1.y, b1.z, b1.w };
    #pragma unroll
    for (int j = 0; j < 2; ++j) {
      const int w = w0 + j * 16 + ncol;
      #pragma unroll
      for (int q = 0; q < 8; ++q) {
        const int oc = ocBase + q;
        out[(size_t)(bIdx * OCH + oc) * WIN + w] = acc[i][j].f[q] + bv[q];
      }
    }
  }
}

extern "C" void kernel_launch(void* const* d_in, const int* in_sizes, int n_in,
                              void* d_out, int out_size, void* d_ws, size_t ws_size,
                              hipStream_t stream) {
  const float* x      = (const float*)d_in[0];
  const float* weight = (const float*)d_in[1];
  const float* offset = (const float*)d_in[2];
  const float* mask   = (const float*)d_in[3];
  const float* bias   = (const float*)d_in[4];
  float* out = (float*)d_out;

  const int blocks = 16 * (WIN / NTILE);                 // B * tilesPerB = 2048
  const size_t wbfBytes = (size_t)OCH * CK * sizeof(unsigned short);  // 384 KB

  if (ws_size >= wbfBytes) {
    // one-time (per call) weight pre-conversion: 98304 packed dwords
    convert_weight_bf16<<<(OCH * CK / 2) / 256, 256, 0, stream>>>(
        weight, (unsigned*)d_ws);
    deform_conv1d_wmma<true><<<blocks, 256, 0, stream>>>(
        x, weight, (const unsigned short*)d_ws, offset, mask, bias, out);
  } else {
    deform_conv1d_wmma<false><<<blocks, 256, 0, stream>>>(
        x, weight, nullptr, offset, mask, bias, out);
  }
}
